// GatedRealLRUBPTT_70824010711758
// MI455X (gfx1250) — compile-verified, matrix operand order checked by reading, not run on hardware
//
#include <hip/hip_runtime.h>
#include <hip/hip_bf16.h>

// ---- types --------------------------------------------------------------
typedef __attribute__((ext_vector_type(16))) __bf16        v16bf;
typedef __attribute__((ext_vector_type(8)))  float         v8f;
typedef __attribute__((ext_vector_type(4)))  unsigned int  uint4v;
typedef __attribute__((ext_vector_type(4)))  float         float4v;
typedef __attribute__((ext_vector_type(4)))  unsigned short ushort4v;

union Frag {
    uint4v u4[2];   // two 16B chunks (bf16 WMMA fragment layout per CDNA5 ISA)
    v16bf  bf;
};

#define T_DIM   4096
#define BATCH   16
#define IN_F    256
#define STATE_F 256
#define ROWS    (T_DIM * BATCH)          // 65536 GEMM rows
#define NCHUNK  64
#define CLEN    64                       // NCHUNK*CLEN == T_DIM
#define CH      (BATCH * STATE_F)        // 4096 chains

__device__ __forceinline__ unsigned short f2bf(float f) {
    unsigned int u = __builtin_bit_cast(unsigned int, f);
    unsigned int r = u + 0x7FFFu + ((u >> 16) & 1u);   // round-to-nearest-even
    return (unsigned short)(r >> 16);
}

// ---- kernel 0: convert the three weight matrices fp32 -> bf16 -----------
// dst layout: [3][STATE_F][IN_F] bf16, K contiguous (NT-GEMM friendly)
__global__ __launch_bounds__(256) void k_cvt_weights(
    const float* __restrict__ w0,   // V_nu  (gate logits)
    const float* __restrict__ w1,   // B     (b_x)
    const float* __restrict__ w2,   // gate_W(gate_x)
    unsigned short* __restrict__ dst)
{
    int gid = blockIdx.x * 256 + threadIdx.x;       // 0 .. 49151 (float4 units)
    const int perMat = (STATE_F * IN_F) / 4;        // 16384
    int mat = gid / perMat;
    int e4  = gid % perMat;
    const float* src = (mat == 0) ? w0 : (mat == 1) ? w1 : w2;
    float4v v = *(const float4v*)(src + e4 * 4);
    ushort4v h = { f2bf(v.x), f2bf(v.y), f2bf(v.z), f2bf(v.w) };
    *(ushort4v*)(dst + (size_t)mat * STATE_F * IN_F + (size_t)e4 * 4) = h;
}

// ---- kernel 1: fused triple GEMM (bf16 WMMA) + gating epilogue ----------
// block = 256 threads (8 waves). Block owns 32 rows x 256 cols x 3 mats.
#define LDS_STRIDE 264   // 256 + 8 ushorts pad -> 528B row stride, bank-rotating
__global__ __launch_bounds__(256) void k_gemm_gate(
    const float* __restrict__ x,             // [ROWS, IN_F] fp32
    const float* __restrict__ nu_log,        // [STATE_F]
    const unsigned short* __restrict__ wB,   // [3][STATE_F][IN_F] bf16
    float* __restrict__ lam_out,             // [ROWS, STATE_F]
    float* __restrict__ u_out)               // [ROWS, STATE_F] (in d_out)
{
    __shared__ unsigned short sA[32 * LDS_STRIDE];

    const int tid = threadIdx.x;
    const long rowBase = (long)blockIdx.x * 32;

    // Stage + convert the 32x256 fp32 A tile into LDS as bf16 (x read once).
#pragma unroll
    for (int i = 0; i < 8; ++i) {
        int idx = tid + i * 256;          // 0..2047 float4 slots
        int r   = idx >> 6;               // row within tile
        int kc  = (idx & 63) << 2;        // K offset (multiple of 4)
        float4v v = *(const float4v*)(x + (rowBase + r) * IN_F + kc);
        ushort4v h = { f2bf(v.x), f2bf(v.y), f2bf(v.z), f2bf(v.w) };
        *(ushort4v*)(sA + r * LDS_STRIDE + kc) = h;
    }
    __syncthreads();

    const int lane = tid & 31, wave = tid >> 5;
    const int mSub = wave >> 2;               // 0/1 -> rows 0..15 / 16..31
    const int nBase = (wave & 3) * 64;        // 4 N-tiles of 16
    const int half = lane >> 4, lr = lane & 15;

    const unsigned short* aRow = sA + (mSub * 16 + lr) * LDS_STRIDE;

    v8f acc[3][4] = {};                       // 96 VGPRs of accumulators

    for (int k0 = 0; k0 < IN_F; k0 += 32) {
        Frag a;
        // CDNA5 16-bit A layout: lanes0-15 K=[k0..k0+7],[k0+16..k0+23];
        //                        lanes16-31 shifted by 8.
        a.u4[0] = *(const uint4v*)(aRow + k0 + half * 8);
        a.u4[1] = *(const uint4v*)(aRow + k0 + 16 + half * 8);
#pragma unroll
        for (int mat = 0; mat < 3; ++mat) {
#pragma unroll
            for (int j = 0; j < 4; ++j) {
                const unsigned short* bp =
                    wB + ((mat << 16) | ((nBase + j * 16 + lr) << 8))
                       + k0 + half * 8;
                Frag b;
                b.u4[0] = *(const uint4v*)(bp);
                b.u4[1] = *(const uint4v*)(bp + 16);
                acc[mat][j] = __builtin_amdgcn_wmma_f32_16x16x32_bf16(
                    false, a.bf, false, b.bf, (short)0, acc[mat][j],
                    false, false);
            }
        }
    }

    // Epilogue: C/D layout -> row = r + 8*half, col = lr + n0
    const long row0 = rowBase + mSub * 16 + half * 8;
#pragma unroll
    for (int j = 0; j < 4; ++j) {
        int n = nBase + j * 16 + lr;
        float an = 3.0f * __expf(nu_log[n]);            // ALPHA * exp(nu_log)
#pragma unroll
        for (int r = 0; r < 8; ++r) {
            float g  = acc[0][j][r];                    // V_nu logits
            float bx = acc[1][j][r];                    // B
            float gx = acc[2][j][r];                    // gate_W
            float gate = 1.0f / (1.0f + __expf(-g));
            float lam  = __expf(-an * (1.0f - gate));
            float gam  = sqrtf(1.0f + 1e-7f - lam * lam);
            long off = (row0 + r) * STATE_F + n;
            lam_out[off] = lam;
            u_out[off]   = gam * bx * gx;
        }
    }
}

// ---- kernel 2a: per-chunk reduction: P = prod(lam), U = local scan ------
__global__ __launch_bounds__(256) void k_chunk_reduce(
    const float* __restrict__ lam, const float* __restrict__ u,
    float* __restrict__ P, float* __restrict__ U)
{
    int gid   = blockIdx.x * 256 + threadIdx.x;   // 0 .. NCHUNK*CH-1
    int chunk = gid >> 12;                        // /CH
    int c     = gid & (CH - 1);
    long base = (long)chunk * CLEN * CH + c;
    float p = 1.0f, s = 0.0f;
#pragma unroll 4
    for (int i = 0; i < CLEN; ++i) {
        float l  = lam[base + (long)i * CH];
        float uu = u  [base + (long)i * CH];
        s = fmaf(l, s, uu);
        p *= l;
    }
    P[gid] = p;
    U[gid] = s;
}

// ---- kernel 2b: carry scan across chunks (4096 threads) -----------------
__global__ __launch_bounds__(256) void k_chunk_carry(
    const float* __restrict__ P, const float* __restrict__ U,
    float* __restrict__ S)                         // carry-in per chunk
{
    int c = blockIdx.x * 256 + threadIdx.x;        // 0..CH-1
    float st = 0.0f;
    for (int ch = 0; ch < NCHUNK; ++ch) {
        S[ch * CH + c] = st;                       // state entering chunk ch
        st = fmaf(P[ch * CH + c], st, U[ch * CH + c]);
    }
}

// ---- kernel 2c: apply — rescan each chunk with its carry, write output --
__global__ __launch_bounds__(256) void k_chunk_apply(
    const float* __restrict__ lam, const float* __restrict__ S,
    float* __restrict__ out,        // holds u on entry, outputs on exit
    float* __restrict__ final_state)
{
    int gid   = blockIdx.x * 256 + threadIdx.x;
    int chunk = gid >> 12;
    int c     = gid & (CH - 1);
    long base = (long)chunk * CLEN * CH + c;
    float st = S[gid];
    for (int i = 0; i < CLEN; ++i) {
        long idx = base + (long)i * CH;
        float l  = lam[idx];
        float uu = out[idx];
        st = fmaf(l, st, uu);
        out[idx] = st;
    }
    if (chunk == NCHUNK - 1) final_state[c] = st;
}

// ---- launch -------------------------------------------------------------
extern "C" void kernel_launch(void* const* d_in, const int* in_sizes, int n_in,
                              void* d_out, int out_size, void* d_ws, size_t ws_size,
                              hipStream_t stream)
{
    const float* x      = (const float*)d_in[0];   // [T,B,IN_F]
    const float* nu_log = (const float*)d_in[1];   // [STATE_F]
    const float* W_B    = (const float*)d_in[2];   // B
    const float* W_gate = (const float*)d_in[3];   // gate_W
    const float* W_Vnu  = (const float*)d_in[4];   // V_nu

    float* out_seq   = (float*)d_out;                          // [T,B,S]
    float* out_final = out_seq + (size_t)ROWS * STATE_F;       // [B,S]

    // workspace partition
    char* ws = (char*)d_ws;
    float*          ws_lam = (float*)ws;                                   // 64 MB
    size_t off = (size_t)ROWS * STATE_F * sizeof(float);
    unsigned short* ws_wB  = (unsigned short*)(ws + off);                  // 384 KB
    off += (size_t)3 * STATE_F * IN_F * sizeof(unsigned short);
    float* ws_P = (float*)(ws + off); off += (size_t)NCHUNK * CH * sizeof(float);
    float* ws_U = (float*)(ws + off); off += (size_t)NCHUNK * CH * sizeof(float);
    float* ws_S = (float*)(ws + off); off += (size_t)NCHUNK * CH * sizeof(float);

    // 0) weights -> bf16 (L2-resident for the whole GEMM pass)
    k_cvt_weights<<<(3 * STATE_F * IN_F / 4 + 255) / 256, 256, 0, stream>>>(
        W_Vnu, W_B, W_gate, ws_wB);

    // 1) fused triple GEMM + gating epilogue: lam -> ws, u -> d_out
    k_gemm_gate<<<ROWS / 32, 256, 0, stream>>>(
        x, nu_log, ws_wB, ws_lam, out_seq);

    // 2) chunked parallel linear scan over T
    k_chunk_reduce<<<(NCHUNK * CH) / 256, 256, 0, stream>>>(
        ws_lam, out_seq, ws_P, ws_U);
    k_chunk_carry<<<CH / 256, 256, 0, stream>>>(ws_P, ws_U, ws_S);
    k_chunk_apply<<<(NCHUNK * CH) / 256, 256, 0, stream>>>(
        ws_lam, ws_S, out_seq, out_final);
}
